// TemporalEncoderLayer_12068858101760
// MI455X (gfx1250) — matrix operand order repaired and checked by reading, not compiled, not run on hardware
//
#include <hip/hip_runtime.h>

// Problem constants (match reference)
#define BATCH   2
#define SEQ     2048
#define DMODEL  1024
#define NH      16
#define DHEAD   64
#define HW      256            // one-sided window
#define DFF     4096
#define MROWS   (BATCH * SEQ)  // 4096

typedef __attribute__((ext_vector_type(16))) __bf16        v16bf;
typedef __attribute__((ext_vector_type(8)))  float         v8f;
typedef __attribute__((ext_vector_type(8)))  unsigned int  v8u;
typedef __attribute__((ext_vector_type(4)))  unsigned int  v4u;
typedef __attribute__((ext_vector_type(8)))  int           v8i;
typedef __attribute__((ext_vector_type(4)))  int           v4i;

union FragU { v8u u; v16bf b; };

__device__ __forceinline__ unsigned short f2bf(float f) {
  union { float f; unsigned int u; } c; c.f = f;
  unsigned int r = c.u + 0x7FFFu + ((c.u >> 16) & 1u);  // round-to-nearest-even
  return (unsigned short)(r >> 16);
}

__device__ __forceinline__ float gelu_exact(float x) {
  return 0.5f * x * (1.0f + erff(x * 0.7071067811865475f));
}

__device__ __forceinline__ v8f wmma_bf16(v16bf a, v16bf b, v8f c) {
  // D = A(16x32 bf16) * B(32x16 bf16) + C(16x16 f32)
  return __builtin_amdgcn_wmma_f32_16x16x32_bf16(false, a, false, b, (short)0, c,
                                                 false, false);
}

// ---------------------------------------------------------------------------
// Tensor Data Mover: 2-D tile load (bf16, row-major) global -> LDS.
// Descriptor per CDNA5 ISA (D# group 0/1; groups 2/3 zero).
// This toolchain declares the 6-arg builtin: (v4u, v8i, v4i, v4i, v8i, i32).
// ---------------------------------------------------------------------------
__device__ __forceinline__ void tdm_load_2d_bf16(unsigned lds_addr, const void* gaddr,
                                                 unsigned tile_d0, unsigned tile_d1,
                                                 unsigned long long tensor_d0,
                                                 unsigned long long tensor_d1,
                                                 unsigned long long d0_stride_elems) {
  unsigned long long ga = (unsigned long long)(uintptr_t)gaddr;
  v4u g0;
  g0[0] = 1u;                                    // count=1 (valid), user mode
  g0[1] = lds_addr;                              // LDS byte address
  g0[2] = (unsigned)(ga & 0xffffffffu);          // global_addr[31:0]
  g0[3] = (unsigned)((ga >> 32) & 0x01ffffffu)   // global_addr[56:32]
          | (2u << 30);                          // type = 2 ("image")
  v8i g1;
  g1[0] = (int)(1u << 16);                       // workgroup_mask=0, data_size=1 (2B)
  g1[1] = (int)((tensor_d0 & 0xffffu) << 16);    // tensor_dim0[15:0] @ bits 63:48
  g1[2] = (int)(((tensor_d0 >> 16) & 0xffffu) |  // tensor_dim0[31:16]
                ((tensor_d1 & 0xffffu) << 16));  // tensor_dim1[15:0]
  g1[3] = (int)(((tensor_d1 >> 16) & 0xffffu) |  // tensor_dim1[31:16]
                ((tile_d0 & 0xffffu) << 16));    // tile_dim0
  g1[4] = (int)(tile_d1 & 0xffffu);              // tile_dim1 ; tile_dim2 = 0
  g1[5] = (int)(d0_stride_elems & 0xffffffffu);  // tensor_dim0_stride[31:0]
  g1[6] = (int)((d0_stride_elems >> 32) & 0xffffu); // stride[47:32]; dim1_stride lo = 0
  g1[7] = 0;                                     // tensor_dim1_stride hi = 0
  v4i z4 = {0, 0, 0, 0};
  v8i z8 = {0, 0, 0, 0, 0, 0, 0, 0};
  __builtin_amdgcn_tensor_load_to_lds(g0, g1, z4, z4, z8, 0);
}

// ---------------------------------------------------------------------------
// fp32 -> bf16 elementwise cast (weights)
// ---------------------------------------------------------------------------
__global__ __launch_bounds__(256)
void cast_bf16_kernel(const float* __restrict__ in, unsigned short* __restrict__ out, int n) {
  int i = blockIdx.x * 256 + threadIdx.x;
  if (i < n) out[i] = f2bf(in[i]);
}

// ---------------------------------------------------------------------------
// LayerNorm over DMODEL, block per row, output bf16
// ---------------------------------------------------------------------------
__global__ __launch_bounds__(256)
void ln_cast_kernel(const float* __restrict__ x, const float* __restrict__ g,
                    const float* __restrict__ bb, unsigned short* __restrict__ out) {
  const int row = blockIdx.x;
  const int tid = threadIdx.x;
  const float* xr = x + (size_t)row * DMODEL;
  float s = 0.f, s2 = 0.f;
  for (int i = tid; i < DMODEL; i += 256) { float v = xr[i]; s += v; s2 += v * v; }
  __shared__ float rs[256], rq[256];
  rs[tid] = s; rq[tid] = s2; __syncthreads();
  for (int off = 128; off > 0; off >>= 1) {
    if (tid < off) { rs[tid] += rs[tid + off]; rq[tid] += rq[tid + off]; }
    __syncthreads();
  }
  const float mean = rs[0] * (1.0f / DMODEL);
  const float var  = rq[0] * (1.0f / DMODEL) - mean * mean;
  const float inv  = rsqrtf(var + 1e-5f);
  for (int i = tid; i < DMODEL; i += 256)
    out[(size_t)row * DMODEL + i] = f2bf((xr[i] - mean) * inv * g[i] + bb[i]);
}

// ---------------------------------------------------------------------------
// bf16 WMMA GEMM: C = A[M,K] * B[K,N] + bias, block tile 64x128, 8 waves.
// A tile is fetched by the Tensor Data Mover (wave 0 issues, TENSORcnt drain,
// barrier publishes); B tile is cooperatively loaded and stored transposed.
// EPI: 0 = store bf16, 1 = GELU + bf16, 2 = fp32 residual add
// ---------------------------------------------------------------------------
template<int EPI>
__global__ __launch_bounds__(256)
void gemm_bf16_kernel(const unsigned short* __restrict__ A,
                      const unsigned short* __restrict__ Bw,
                      const float* __restrict__ bias,
                      const float* __restrict__ res,
                      float* __restrict__ outf,
                      unsigned short* __restrict__ outb,
                      int M, int N, int K) {
  __shared__ __align__(16) unsigned short lda[64 * 32];    // A tile, row-major [m][k]
  __shared__ __align__(16) unsigned short ldbT[128 * 32];  // B tile transposed [n][k]

  const int tid   = threadIdx.x;
  const int lane  = tid & 31;
  const int l15   = lane & 15;
  const int lhalf = lane >> 4;
  const int wid   = tid >> 5;
  const int wm    = wid >> 2;     // 0..1
  const int wn    = wid & 3;      // 0..3
  const int mbase = blockIdx.y * 64;
  const int nbase = blockIdx.x * 128;
  const unsigned lda_lds = (unsigned)(uintptr_t)(&lda[0]);  // LDS byte address (addr[31:0])

  v8f zero = {};
  v8f acc[2][2];
  acc[0][0] = zero; acc[0][1] = zero; acc[1][0] = zero; acc[1][1] = zero;

  for (int k0 = 0; k0 < K; k0 += 32) {
    // A tile via TDM: tile 32 (contig k) x 64 (rows m), row stride K elems
    if (wid == 0) {
      tdm_load_2d_bf16(lda_lds, A + (size_t)mbase * K + k0,
                       /*tile_d0=*/32, /*tile_d1=*/64,
                       /*tensor_d0=*/(unsigned long long)K,
                       /*tensor_d1=*/64ull,
                       /*stride=*/(unsigned long long)K);
    }
    // Prefetch next B tile while this one is consumed
    if (k0 + 32 < K)
      __builtin_prefetch(Bw + (size_t)(k0 + 32) * N + nbase + (tid & 15) * 8, 0, 1);
    #pragma unroll
    for (int pass = 0; pass < 2; ++pass) { // B: 32x128, stored transposed
      int row = (tid >> 4) + pass * 16;    // k
      int c8  = (tid & 15) * 8;            // n
      int4 val = *reinterpret_cast<const int4*>(Bw + (size_t)(k0 + row) * N + nbase + c8);
      const unsigned short* e = reinterpret_cast<const unsigned short*>(&val);
      #pragma unroll
      for (int i = 0; i < 8; ++i) ldbT[(c8 + i) * 32 + row] = e[i];
    }
    if (wid == 0) __builtin_amdgcn_s_wait_tensorcnt(0);  // A tile landed in LDS
    __syncthreads();

    v16bf afr[2], bfr[2];
    #pragma unroll
    for (int mt = 0; mt < 2; ++mt) {  // A fragment per ISA 16-bit A layout
      FragU fu;
      int m = wm * 32 + mt * 16 + l15;
      int khalf = lhalf * 8;
      #pragma unroll
      for (int vv = 0; vv < 8; ++vv) {
        int kk = (vv < 4 ? 2 * vv : 16 + 2 * (vv - 4)) + khalf;
        fu.u[vv] = *reinterpret_cast<const unsigned int*>(&lda[m * 32 + kk]);
      }
      afr[mt] = fu.b;
    }
    #pragma unroll
    for (int nt = 0; nt < 2; ++nt) {  // B fragment: pairs along K contiguous in ldbT
      FragU fu;
      int n = wn * 32 + nt * 16 + l15;
      int khalf = lhalf * 8;
      #pragma unroll
      for (int vv = 0; vv < 8; ++vv)
        fu.u[vv] = *reinterpret_cast<const unsigned int*>(&ldbT[n * 32 + khalf + (vv < 4 ? 2 * vv : 16 + 2 * (vv - 4))]);
      bfr[nt] = fu.b;
    }
    #pragma unroll
    for (int mt = 0; mt < 2; ++mt)
      #pragma unroll
      for (int nt = 0; nt < 2; ++nt)
        acc[mt][nt] = wmma_bf16(afr[mt], bfr[nt], acc[mt][nt]);
    __syncthreads();
  }

  #pragma unroll
  for (int mt = 0; mt < 2; ++mt)
    #pragma unroll
    for (int nt = 0; nt < 2; ++nt)
      #pragma unroll
      for (int r = 0; r < 8; ++r) {
        int grow = mbase + wm * 32 + mt * 16 + lhalf * 8 + r;
        int gcol = nbase + wn * 32 + nt * 16 + l15;
        float val = acc[mt][nt][r] + bias[gcol];
        size_t off = (size_t)grow * N + gcol;
        if (EPI == 0)      outb[off] = f2bf(val);
        else if (EPI == 1) outb[off] = f2bf(gelu_exact(val));
        else               outf[off] = res[off] + val;
      }
}

// ---------------------------------------------------------------------------
// Flash-style sliding-window + global attention.
// Block = (b, h, 128 queries); 8 waves x 16-query tiles; 32-key chunks.
// ---------------------------------------------------------------------------
__global__ __launch_bounds__(256)
void attn_kernel(const unsigned short* __restrict__ q,
                 const unsigned short* __restrict__ k,
                 const unsigned short* __restrict__ v,
                 const int* __restrict__ isg,
                 unsigned short* __restrict__ ctx) {
  const int tid   = threadIdx.x;
  const int lane  = tid & 31;
  const int l15   = lane & 15;
  const int lhalf = lane >> 4;
  const int w     = tid >> 5;

  const int bid = blockIdx.x;
  const int b   = bid / (NH * (SEQ / 128));
  const int rem = bid % (NH * (SEQ / 128));
  const int h   = rem >> 4;
  const int qb0 = (rem & 15) * 128;
  const int qbase = qb0 + w * 16;

  __shared__ __align__(16) unsigned short kt[32 * 64];       // K chunk [key][d]
  __shared__ __align__(16) unsigned short vt[64 * 32];       // V chunk transposed [d][key]
  __shared__ __align__(16) unsigned short pst[8][16 * 32];   // per-wave P staging
  __shared__ int s_qany;

  if (tid == 0) s_qany = 0;
  __syncthreads();
  if (tid < 128 && isg[b * SEQ + qb0 + tid]) atomicOr(&s_qany, 1);
  __syncthreads();
  const int qany = s_qany;

  int gq[8];
  #pragma unroll
  for (int r = 0; r < 8; ++r) gq[r] = isg[b * SEQ + qbase + lhalf * 8 + r];

  // Q fragments (A layout), DH=64 -> two 32-wide K chunks
  v16bf afq[2];
  {
    const size_t base = ((size_t)(b * SEQ + qbase + l15)) * DMODEL + h * DHEAD;
    const int khalf = lhalf * 8;
    #pragma unroll
    for (int dc = 0; dc < 2; ++dc) {
      FragU fu;
      #pragma unroll
      for (int vv = 0; vv < 8; ++vv) {
        int kk = (vv < 4 ? 2 * vv : 16 + 2 * (vv - 4)) + khalf;
        fu.u[vv] = *reinterpret_cast<const unsigned int*>(&q[base + dc * 32 + kk]);
      }
      afq[dc] = fu.b;
    }
  }

  float mi[8], li[8];
  v8f zero = {};
  v8f acc[4];
  #pragma unroll
  for (int r = 0; r < 8; ++r) { mi[r] = -3.0e38f; li[r] = 0.f; }
  #pragma unroll
  for (int nt = 0; nt < 4; ++nt) acc[nt] = zero;

  const float scale = 0.125f;  // 1/sqrt(DH)

  for (int c = 0; c < SEQ / 32; ++c) {
    const int kbase = c * 32;
    const int gk = isg[b * SEQ + kbase + lane];
    const unsigned long long bal = __ballot(gk != 0);
    const bool kany = (bal != 0ull);
    const bool overlap = (kbase + 31 >= qb0 - HW) && (kbase <= qb0 + 127 + HW);
    if (!overlap && !kany && !qany) continue;  // block-uniform skip

    { // cooperative K / V tile loads
      int key = tid >> 3;          // 0..31
      int c8  = (tid & 7) * 8;     // 0..56
      size_t src = ((size_t)(b * SEQ + kbase + key)) * DMODEL + h * DHEAD + c8;
      int4 kr = *reinterpret_cast<const int4*>(k + src);
      *reinterpret_cast<int4*>(&kt[key * 64 + c8]) = kr;
      int4 vr = *reinterpret_cast<const int4*>(v + src);
      const unsigned short* e = reinterpret_cast<const unsigned short*>(&vr);
      #pragma unroll
      for (int i = 0; i < 8; ++i) vt[(c8 + i) * 32 + key] = e[i];
    }
    __syncthreads();

    // scores: Q(16x64) * K^T -> two 16x16 tiles
    v8f st[2]; st[0] = zero; st[1] = zero;
    #pragma unroll
    for (int sub = 0; sub < 2; ++sub)
      #pragma unroll
      for (int dc = 0; dc < 2; ++dc) {
        FragU fu;
        int n = sub * 16 + l15;            // key column
        int kh = lhalf * 8;
        #pragma unroll
        for (int vv = 0; vv < 8; ++vv)
          fu.u[vv] = *reinterpret_cast<const unsigned int*>(&kt[n * 64 + dc * 32 + kh + (vv < 4 ? 2 * vv : 16 + 2 * (vv - 4))]);
        st[sub] = wmma_bf16(afq[dc], fu.b, st[sub]);
      }

    int gkc[2];
    gkc[0] = __shfl(gk, l15);
    gkc[1] = __shfl(gk, 16 + l15);

    float pv[2][8];
    #pragma unroll
    for (int r = 0; r < 8; ++r) {
      const int qi = qbase + lhalf * 8 + r;
      float sv[2], rm = -3.0e38f;
      #pragma unroll
      for (int sub = 0; sub < 2; ++sub) {
        int ki = kbase + sub * 16 + l15;
        int d = qi - ki;
        bool allowed = (d <= HW && d >= -HW) || gq[r] || gkc[sub];
        float s = allowed ? st[sub][r] * scale : -1.0e9f;
        sv[sub] = s;
        rm = fmaxf(rm, s);
      }
      #pragma unroll
      for (int off = 1; off < 16; off <<= 1) rm = fmaxf(rm, __shfl_xor(rm, off));
      float newm = fmaxf(mi[r], rm);
      float sf = __expf(mi[r] - newm);
      mi[r] = newm;
      float p0 = __expf(sv[0] - newm);
      float p1 = __expf(sv[1] - newm);
      float ls = p0 + p1;
      #pragma unroll
      for (int off = 1; off < 16; off <<= 1) ls += __shfl_xor(ls, off);
      li[r] = li[r] * sf + ls;
      #pragma unroll
      for (int nt = 0; nt < 4; ++nt) acc[nt][r] *= sf;
      pv[0][r] = p0; pv[1][r] = p1;
    }

    // stage P (C layout) -> reload as A fragment (wave-private LDS)
    #pragma unroll
    for (int sub = 0; sub < 2; ++sub)
      #pragma unroll
      for (int r = 0; r < 8; ++r)
        pst[w][(lhalf * 8 + r) * 32 + sub * 16 + l15] = f2bf(pv[sub][r]);

    v16bf ap;
    {
      FragU fu;
      int khalf = lhalf * 8;
      #pragma unroll
      for (int vv = 0; vv < 8; ++vv) {
        int kk = (vv < 4 ? 2 * vv : 16 + 2 * (vv - 4)) + khalf;
        fu.u[vv] = *reinterpret_cast<const unsigned int*>(&pst[w][l15 * 32 + kk]);
      }
      ap = fu.b;
    }

    // ctx += P(16x32) * V(32x64)
    #pragma unroll
    for (int nt = 0; nt < 4; ++nt) {
      FragU fu;
      int n = nt * 16 + l15;               // d column
      int kh = lhalf * 8;
      #pragma unroll
      for (int vv = 0; vv < 8; ++vv)
        fu.u[vv] = *reinterpret_cast<const unsigned int*>(&vt[n * 32 + kh + (vv < 4 ? 2 * vv : 16 + 2 * (vv - 4))]);
      acc[nt] = wmma_bf16(ap, fu.b, acc[nt]);
    }
    __syncthreads();
  }

  // normalize and write ctx (bf16)
  #pragma unroll
  for (int r = 0; r < 8; ++r) {
    float inv = 1.0f / li[r];  // self is always in-window -> li > 0
    int qi = qbase + lhalf * 8 + r;
    size_t base = ((size_t)(b * SEQ + qi)) * DMODEL + h * DHEAD;
    #pragma unroll
    for (int nt = 0; nt < 4; ++nt)
      ctx[base + nt * 16 + l15] = f2bf(acc[nt][r] * inv);
  }
}

// ---------------------------------------------------------------------------
extern "C" void kernel_launch(void* const* d_in, const int* in_sizes, int n_in,
                              void* d_out, int out_size, void* d_ws, size_t ws_size,
                              hipStream_t stream) {
  const float* x   = (const float*)d_in[0];
  const int*   isg = (const int*)d_in[1];
  const float* Wq  = (const float*)d_in[2];  const float* bq = (const float*)d_in[3];
  const float* Wk  = (const float*)d_in[4];  const float* bk = (const float*)d_in[5];
  const float* Wv  = (const float*)d_in[6];  const float* bv = (const float*)d_in[7];
  const float* Wo  = (const float*)d_in[8];  const float* bo = (const float*)d_in[9];
  const float* g1  = (const float*)d_in[10]; const float* be1 = (const float*)d_in[11];
  const float* g2  = (const float*)d_in[12]; const float* be2 = (const float*)d_in[13];
  const float* W1  = (const float*)d_in[14]; const float* b1 = (const float*)d_in[15];
  const float* W2  = (const float*)d_in[16]; const float* b2 = (const float*)d_in[17];

  char* ws = (char*)d_ws;
  const size_t MB = 1024ull * 1024ull;
  unsigned short* hb  = (unsigned short*)(ws + 0 * MB);    // [4096,1024] bf16
  unsigned short* qb  = (unsigned short*)(ws + 8 * MB);
  unsigned short* kb  = (unsigned short*)(ws + 16 * MB);
  unsigned short* vb  = (unsigned short*)(ws + 24 * MB);
  unsigned short* ctb = (unsigned short*)(ws + 32 * MB);
  float*          xm  = (float*)(ws + 40 * MB);            // [4096,1024] f32
  unsigned short* h2b = (unsigned short*)(ws + 56 * MB);
  unsigned short* ffb = (unsigned short*)(ws + 64 * MB);   // [4096,4096] bf16
  unsigned short* Wqb = (unsigned short*)(ws + 96 * MB);
  unsigned short* Wkb = (unsigned short*)(ws + 98 * MB);
  unsigned short* Wvb = (unsigned short*)(ws + 100 * MB);
  unsigned short* Wob = (unsigned short*)(ws + 102 * MB);
  unsigned short* W1b = (unsigned short*)(ws + 104 * MB);
  unsigned short* W2b = (unsigned short*)(ws + 112 * MB);

  const int nDD = DMODEL * DMODEL;        // 1048576
  const int nDF = DMODEL * DFF;           // 4194304
  cast_bf16_kernel<<<(nDD + 255) / 256, 256, 0, stream>>>(Wq, Wqb, nDD);
  cast_bf16_kernel<<<(nDD + 255) / 256, 256, 0, stream>>>(Wk, Wkb, nDD);
  cast_bf16_kernel<<<(nDD + 255) / 256, 256, 0, stream>>>(Wv, Wvb, nDD);
  cast_bf16_kernel<<<(nDD + 255) / 256, 256, 0, stream>>>(Wo, Wob, nDD);
  cast_bf16_kernel<<<(nDF + 255) / 256, 256, 0, stream>>>(W1, W1b, nDF);
  cast_bf16_kernel<<<(nDF + 255) / 256, 256, 0, stream>>>(W2, W2b, nDF);

  // LN1
  ln_cast_kernel<<<MROWS, 256, 0, stream>>>(x, g1, be1, hb);

  // Q/K/V projections (bf16 out)
  dim3 gDD(DMODEL / 128, MROWS / 64);
  gemm_bf16_kernel<0><<<gDD, 256, 0, stream>>>(hb, Wqb, bq, nullptr, nullptr, qb,
                                               MROWS, DMODEL, DMODEL);
  gemm_bf16_kernel<0><<<gDD, 256, 0, stream>>>(hb, Wkb, bk, nullptr, nullptr, kb,
                                               MROWS, DMODEL, DMODEL);
  gemm_bf16_kernel<0><<<gDD, 256, 0, stream>>>(hb, Wvb, bv, nullptr, nullptr, vb,
                                               MROWS, DMODEL, DMODEL);

  // Attention
  attn_kernel<<<BATCH * NH * (SEQ / 128), 256, 0, stream>>>(qb, kb, vb, isg, ctb);

  // Output projection + residual (fp32)
  gemm_bf16_kernel<2><<<gDD, 256, 0, stream>>>(ctb, Wob, bo, x, xm, nullptr,
                                               MROWS, DMODEL, DMODEL);
  // LN2
  ln_cast_kernel<<<MROWS, 256, 0, stream>>>(xm, g2, be2, h2b);

  // MLP up (GELU, bf16 out)
  dim3 gDF(DFF / 128, MROWS / 64);
  gemm_bf16_kernel<1><<<gDF, 256, 0, stream>>>(h2b, W1b, b1, nullptr, nullptr, ffb,
                                               MROWS, DFF, DMODEL);
  // MLP down + residual -> d_out (fp32)
  gemm_bf16_kernel<2><<<gDD, 256, 0, stream>>>(ffb, W2b, b2, xm, (float*)d_out, nullptr,
                                               MROWS, DMODEL, DFF);
}
